// Selformer_59124519797205
// MI455X (gfx1250) — compile-verified, matrix operand order checked by reading, not run on hardware
//
#include <hip/hip_runtime.h>
#include <math.h>

#define HEADS 16

typedef __attribute__((ext_vector_type(16))) _Float16 v16h;
typedef __attribute__((ext_vector_type(8)))  float    v8f;
typedef __attribute__((ext_vector_type(4)))  unsigned u32x4;

union AV { v16h v; unsigned u[8]; _Float16 h[16]; };
union HV { v16h v; u32x4 q[2]; };

__device__ __forceinline__ v8f v8f_zero() {
    v8f z;
#pragma unroll
    for (int i = 0; i < 8; ++i) z[i] = 0.0f;
    return z;
}

__device__ __forceinline__ float gelu_tanh(float x) {
    const float k0 = 0.7978845608028654f; // sqrt(2/pi)
    const float k1 = 0.044715f;
    return 0.5f * x * (1.0f + tanhf(k0 * (x + k1 * x * x * x)));
}

// CDNA5 async global->LDS copy (ASYNCcnt-tracked), one 16B chunk per lane.
__device__ __forceinline__ void async_cp16(unsigned lds_off, const _Float16* g) {
    asm volatile("global_load_async_to_lds_b128 %0, %1, off"
                 :: "v"(lds_off), "v"((unsigned long long)(uintptr_t)g) : "memory");
}

// ---------------------------------------------------------------------------
// Convert f32 weight (K x N, row-major) -> f16 transposed (N x K, row-major)
// so WMMA B-operand (and async LDS staging) loads are K-contiguous.
// ---------------------------------------------------------------------------
__global__ void cvt_t_f16(const float* __restrict__ in, _Float16* __restrict__ out,
                          int K, int N) {
    size_t idx = (size_t)blockIdx.x * blockDim.x + threadIdx.x;
    size_t tot = (size_t)K * N;
    if (idx >= tot) return;
    size_t nn = idx / K;
    size_t kk = idx - nn * K;
    out[idx] = (_Float16)in[kk * (size_t)N + nn];
}

// f16 (R x C) -> f16 transposed (C x R):  out[c*R + r] = in[r*C + c]
__global__ void transpose_f16(const _Float16* __restrict__ in, _Float16* __restrict__ out,
                              int R, int C) {
    size_t idx = (size_t)blockIdx.x * blockDim.x + threadIdx.x;
    size_t tot = (size_t)R * C;
    if (idx >= tot) return;
    size_t c = idx / R;
    size_t r = idx - c * R;
    out[idx] = in[r * (size_t)C + c];
}

// ---------------------------------------------------------------------------
// Row LayerNorm, f32 in -> f16 out.  One 256-thread block per row.
// ---------------------------------------------------------------------------
__global__ void ln_f16_kernel(const float* __restrict__ x, const float* __restrict__ w,
                              const float* __restrict__ b, _Float16* __restrict__ out,
                              int d) {
    int row = blockIdx.x;
    const float* xr = x + (size_t)row * d;
    float s = 0.f, ss = 0.f;
    for (int i = threadIdx.x; i < d; i += blockDim.x) {
        float v = xr[i];
        s += v; ss += v * v;
    }
    __shared__ float rs[256], rss[256];
    rs[threadIdx.x] = s; rss[threadIdx.x] = ss;
    __syncthreads();
    for (int st = 128; st > 0; st >>= 1) {
        if (threadIdx.x < st) { rs[threadIdx.x] += rs[threadIdx.x + st];
                                rss[threadIdx.x] += rss[threadIdx.x + st]; }
        __syncthreads();
    }
    float mean = rs[0] / (float)d;
    float var  = rss[0] / (float)d - mean * mean;
    float inv  = rsqrtf(var + 1e-12f);
    for (int i = threadIdx.x; i < d; i += blockDim.x)
        out[(size_t)row * d + i] = (_Float16)((xr[i] - mean) * inv * w[i] + b[i]);
}

// ---------------------------------------------------------------------------
// Tiled WMMA GEMM with async global->LDS double-buffered staging.
//   out = act(A(MxK,f16) @ Bt(NxK,f16)^T + bias) (+ resid)
// Block: 256 threads (8 waves) computes 128x128 tile.
// Wave (wm in 0..1, wn in 0..3) computes 64x32 = 4x2 WMMA tiles.
// K staged in 32-wide slabs; sA/sB rows are 64B.  Copies use
// global_load_async_to_lds_b128 with s_wait_asynccnt<=4 pipelining.
// ---------------------------------------------------------------------------
__global__ __launch_bounds__(256)
void wmma_gemm_tiled(const _Float16* __restrict__ A, const _Float16* __restrict__ Bt,
                     const float* __restrict__ bias, const float* __restrict__ resid,
                     float* __restrict__ outf, _Float16* __restrict__ outh,
                     int M, int N, int K, int act) {
    __shared__ _Float16 sA[2][128 * 32];
    __shared__ _Float16 sB[2][128 * 32];

    const int tid  = threadIdx.x;
    const int lane = tid & 31;
    const int wave = tid >> 5;
    const int col  = lane & 15;   // M (A) / N (B,C) index within 16-tile
    const int hh   = lane >> 4;   // lane-half -> +8 K offset in A/B operands
    const int wn   = wave & 3;    // 4 N-groups of 32
    const int wm   = wave >> 2;   // 2 M-groups of 64
    const int m0   = blockIdx.y * 128;
    const int n0   = blockIdx.x * 128;

    const unsigned sAbase = (unsigned)(uintptr_t)(&sA[0][0]);
    const unsigned sBbase = (unsigned)(uintptr_t)(&sB[0][0]);

    // per-thread chunk assignment: rows r0 and r0+64, 16B chunk qc of the 64B row
    const int r0 = tid >> 2;
    const int qc = tid & 3;

    // stage one 128x32-half slab of A and B into LDS buffer `buf` (straight-line)
    auto stage = [&](int buf, int kk) {
        unsigned lbuf = (unsigned)(buf * 8192 + qc * 16);
        async_cp16(sAbase + lbuf + r0 * 64,
                   A + (size_t)(m0 + r0) * K + kk + qc * 8);
        async_cp16(sAbase + lbuf + (r0 + 64) * 64,
                   A + (size_t)(m0 + r0 + 64) * K + kk + qc * 8);
        async_cp16(sBbase + lbuf + r0 * 64,
                   Bt + (size_t)(n0 + r0) * K + kk + qc * 8);
        async_cp16(sBbase + lbuf + (r0 + 64) * 64,
                   Bt + (size_t)(n0 + r0 + 64) * K + kk + qc * 8);
    };

    v8f acc[4][2];
#pragma unroll
    for (int mi = 0; mi < 4; ++mi)
#pragma unroll
        for (int ni = 0; ni < 2; ++ni) acc[mi][ni] = v8f_zero();

    const int ksteps = K >> 5;
    stage(0, 0);
    for (int step = 0; step < ksteps; ++step) {
        const int buf = step & 1;
        if (step + 1 < ksteps) {
            stage(buf ^ 1, (step + 1) << 5);
            // 4 async ops just issued for next stage; in-order completion =>
            // waiting to <=4 guarantees the current stage has landed.
            asm volatile("s_wait_asynccnt 4" ::: "memory");
        } else {
            asm volatile("s_wait_asynccnt 0" ::: "memory");
        }
        __syncthreads();

        HV a[4], b[2];
#pragma unroll
        for (int mi = 0; mi < 4; ++mi) {
            const _Float16* p = &sA[buf][(wm * 64 + mi * 16 + col) * 32 + 8 * hh];
            a[mi].q[0] = *(const u32x4*)(p);
            a[mi].q[1] = *(const u32x4*)(p + 16);
        }
#pragma unroll
        for (int ni = 0; ni < 2; ++ni) {
            const _Float16* p = &sB[buf][(wn * 32 + ni * 16 + col) * 32 + 8 * hh];
            b[ni].q[0] = *(const u32x4*)(p);
            b[ni].q[1] = *(const u32x4*)(p + 16);
        }
#pragma unroll
        for (int mi = 0; mi < 4; ++mi)
#pragma unroll
            for (int ni = 0; ni < 2; ++ni)
                acc[mi][ni] = __builtin_amdgcn_wmma_f32_16x16x32_f16(
                    false, a[mi].v, false, b[ni].v, (short)0, acc[mi][ni], false, false);
        __syncthreads();   // protect LDS buffers before next async issue
    }

    float bn0 = bias ? bias[n0 + wn * 32 + col]      : 0.0f;
    float bn1 = bias ? bias[n0 + wn * 32 + 16 + col] : 0.0f;
#pragma unroll
    for (int mi = 0; mi < 4; ++mi) {
#pragma unroll
        for (int ni = 0; ni < 2; ++ni) {
            float bn = ni ? bn1 : bn0;
#pragma unroll
            for (int r = 0; r < 8; ++r) {
                float val = acc[mi][ni][r] + bn;
                if (act == 1) val = gelu_tanh(val);
                size_t idx = (size_t)(m0 + wm * 64 + mi * 16 + r + 8 * hh) * N
                           + (n0 + wn * 32 + ni * 16 + col);
                if (resid) val += resid[idx];
                if (outf) outf[idx] = val;
                if (outh) outh[idx] = (_Float16)val;
            }
        }
    }
}

// ---------------------------------------------------------------------------
// Causal flash attention.  One wave per (16-query block, head).
// q/k: (n x d) f16, head-h slice at column h*64, dh=64.
// vT: (d x n) f16 transposed V so the P@V B-operand is K-contiguous.
// Keys processed 32 at a time: S = Q Kb^T (4 WMMAs), online softmax,
// P restaged via LDS (C-layout -> A-layout), O += P Vb (4 WMMAs).
// ---------------------------------------------------------------------------
__global__ void flash_attn(const _Float16* __restrict__ q, const _Float16* __restrict__ k,
                           const _Float16* __restrict__ vT, _Float16* __restrict__ o,
                           int n, int d) {
    const int dh = 64;
    const float scale = 0.125f;              // 1/sqrt(64)
    int lane = threadIdx.x & 31;
    int col  = lane & 15;
    int hh   = lane >> 4;
    int head = blockIdx.y;
    int q0   = blockIdx.x * 16;
    const size_t hoff = (size_t)head * dh;

    __shared__ _Float16 sP[16][32];

    AV aq0, aq1;
    const _Float16* qrow = q + (size_t)(q0 + col) * d + hoff;
#pragma unroll
    for (int j = 0; j < 8; ++j) {
        int ko = 2 * j + 8 * hh + ((j >= 4) ? 8 : 0);
        aq0.u[j] = *(const unsigned*)(qrow + ko);
        aq1.u[j] = *(const unsigned*)(qrow + 32 + ko);
    }

    v8f acc[4];
#pragma unroll
    for (int t = 0; t < 4; ++t) acc[t] = v8f_zero();
    float mrow[8], lrow[8];
#pragma unroll
    for (int r = 0; r < 8; ++r) { mrow[r] = -1e30f; lrow[r] = 0.0f; }

    int nkb = (q0 + 16 + 31) / 32;
    for (int kb = 0; kb < nkb; ++kb) {
        int k0 = kb * 32;
        // ---- S = Q Kb^T : two 16x16 score tiles
        AV b0, b1, b2, b3;
        const _Float16* kr0 = k + (size_t)(k0 + col) * d + hoff;
        const _Float16* kr1 = k + (size_t)(k0 + 16 + col) * d + hoff;
#pragma unroll
        for (int j = 0; j < 8; ++j) {
            int ko = 2 * j + 8 * hh + ((j >= 4) ? 8 : 0);
            b0.u[j] = *(const unsigned*)(kr0 + ko);
            b1.u[j] = *(const unsigned*)(kr0 + 32 + ko);
            b2.u[j] = *(const unsigned*)(kr1 + ko);
            b3.u[j] = *(const unsigned*)(kr1 + 32 + ko);
        }
        v8f s0 = v8f_zero(), s1 = v8f_zero();
        s0 = __builtin_amdgcn_wmma_f32_16x16x32_f16(false, aq0.v, false, b0.v, (short)0, s0, false, false);
        s0 = __builtin_amdgcn_wmma_f32_16x16x32_f16(false, aq1.v, false, b1.v, (short)0, s0, false, false);
        s1 = __builtin_amdgcn_wmma_f32_16x16x32_f16(false, aq0.v, false, b2.v, (short)0, s1, false, false);
        s1 = __builtin_amdgcn_wmma_f32_16x16x32_f16(false, aq1.v, false, b3.v, (short)0, s1, false, false);

        // ---- scale + causal mask + online softmax (rows r+8*hh, cols lane%16)
        float rmax[8];
#pragma unroll
        for (int r = 0; r < 8; ++r) {
            int qg = q0 + r + 8 * hh;
            float a0 = s0[r] * scale;
            float a1 = s1[r] * scale;
            if (k0 + col > qg)      a0 = -1e30f;
            if (k0 + 16 + col > qg) a1 = -1e30f;
            s0[r] = a0; s1[r] = a1;
            float mx = fmaxf(a0, a1);
#pragma unroll
            for (int off = 1; off < 16; off <<= 1)
                mx = fmaxf(mx, __shfl_xor(mx, off, 32));
            rmax[r] = mx;
        }
#pragma unroll
        for (int r = 0; r < 8; ++r) {
            float mn = fmaxf(mrow[r], rmax[r]);
            float alpha = __expf(mrow[r] - mn);
            float p0 = __expf(s0[r] - mn);
            float p1 = __expf(s1[r] - mn);
            float rsum = p0 + p1;
#pragma unroll
            for (int off = 1; off < 16; off <<= 1)
                rsum += __shfl_xor(rsum, off, 32);
            lrow[r] = lrow[r] * alpha + rsum;
            mrow[r] = mn;
#pragma unroll
            for (int t = 0; t < 4; ++t) acc[t][r] *= alpha;
            sP[r + 8 * hh][col]      = (_Float16)p0;
            sP[r + 8 * hh][16 + col] = (_Float16)p1;
        }
        __syncthreads();

        // ---- P as A-operand (16x32) from LDS
        AV ap;
#pragma unroll
        for (int j = 0; j < 8; ++j) {
            int ko = 2 * j + 8 * hh + ((j >= 4) ? 8 : 0);
            ap.u[j] = *(const unsigned*)(&sP[col][ko]);
        }
        // ---- O += P @ Vb ; vT makes each B-operand two b128 loads
#pragma unroll
        for (int t = 0; t < 4; ++t) {
            HV bv;
            const _Float16* vb = vT + (size_t)(hoff + t * 16 + col) * n + k0 + 8 * hh;
            bv.q[0] = *(const u32x4*)(vb);
            bv.q[1] = *(const u32x4*)(vb + 16);
            acc[t] = __builtin_amdgcn_wmma_f32_16x16x32_f16(false, ap.v, false, bv.v,
                                                            (short)0, acc[t], false, false);
        }
        __syncthreads();
    }

#pragma unroll
    for (int t = 0; t < 4; ++t) {
#pragma unroll
        for (int r = 0; r < 8; ++r) {
            float val = acc[t][r] / lrow[r];
            o[(size_t)(q0 + r + 8 * hh) * d + hoff + t * 16 + col] = (_Float16)val;
        }
    }
}

// ---------------------------------------------------------------------------
// Host-side orchestration
// ---------------------------------------------------------------------------
extern "C" void kernel_launch(void* const* d_in, const int* in_sizes, int n_in,
                              void* d_out, int out_size, void* d_ws, size_t ws_size,
                              hipStream_t stream) {
    const int d   = in_sizes[1];          // 1024 (ln1_w)
    const int n   = in_sizes[0] / d;      // 4096
    const int dff = in_sizes[14];         // 4096 (bi)

    const float* x     = (const float*)d_in[0];
    const float* ln1_w = (const float*)d_in[1];
    const float* ln1_b = (const float*)d_in[2];
    const float* Wq    = (const float*)d_in[3];
    const float* bq    = (const float*)d_in[4];
    const float* Wk    = (const float*)d_in[5];
    const float* bk    = (const float*)d_in[6];
    const float* Wv    = (const float*)d_in[7];
    const float* bv    = (const float*)d_in[8];
    const float* Wo    = (const float*)d_in[9];
    const float* bo    = (const float*)d_in[10];
    const float* ln2_w = (const float*)d_in[11];
    const float* ln2_b = (const float*)d_in[12];
    const float* Wi    = (const float*)d_in[13];
    const float* bi    = (const float*)d_in[14];
    const float* Wout  = (const float*)d_in[15];
    const float* bout  = (const float*)d_in[16];

    char* ws = (char*)d_ws;
    size_t off = 0;
    auto alloc = [&](size_t bytes) -> void* {
        void* p = ws + off;
        off += (bytes + 255) & ~(size_t)255;
        return p;
    };
    _Float16* WqT   = (_Float16*)alloc((size_t)d * d * 2);
    _Float16* WkT   = (_Float16*)alloc((size_t)d * d * 2);
    _Float16* WvT   = (_Float16*)alloc((size_t)d * d * 2);
    _Float16* WoT   = (_Float16*)alloc((size_t)d * d * 2);
    _Float16* WiT   = (_Float16*)alloc((size_t)d * dff * 2);
    _Float16* WoutT = (_Float16*)alloc((size_t)dff * d * 2);
    _Float16* h1    = (_Float16*)alloc((size_t)n * d * 2);
    _Float16* q16   = (_Float16*)alloc((size_t)n * d * 2);
    _Float16* k16   = (_Float16*)alloc((size_t)n * d * 2);
    _Float16* v16   = (_Float16*)alloc((size_t)n * d * 2);
    _Float16* vTT   = (_Float16*)alloc((size_t)n * d * 2);
    _Float16* ao16  = (_Float16*)alloc((size_t)n * d * 2);
    float*    x1    = (float*)   alloc((size_t)n * d * 4);
    _Float16* h2    = (_Float16*)alloc((size_t)n * d * 2);
    _Float16* m16   = (_Float16*)alloc((size_t)n * dff * 2);

    const int T = 256;
    size_t wdd = (size_t)d * d, wdf = (size_t)d * dff, wnd = (size_t)n * d;
    cvt_t_f16<<<(unsigned)((wdd + T - 1) / T), T, 0, stream>>>(Wq, WqT, d, d);
    cvt_t_f16<<<(unsigned)((wdd + T - 1) / T), T, 0, stream>>>(Wk, WkT, d, d);
    cvt_t_f16<<<(unsigned)((wdd + T - 1) / T), T, 0, stream>>>(Wv, WvT, d, d);
    cvt_t_f16<<<(unsigned)((wdd + T - 1) / T), T, 0, stream>>>(Wo, WoT, d, d);
    cvt_t_f16<<<(unsigned)((wdf + T - 1) / T), T, 0, stream>>>(Wi, WiT, d, dff);
    cvt_t_f16<<<(unsigned)((wdf + T - 1) / T), T, 0, stream>>>(Wout, WoutT, dff, d);

    // LN1 -> h1 (f16)
    ln_f16_kernel<<<n, 256, 0, stream>>>(x, ln1_w, ln1_b, h1, d);

    // QKV projections (f16 out)
    dim3 gdd(d / 128, n / 128);
    wmma_gemm_tiled<<<gdd, 256, 0, stream>>>(h1, WqT, bq, nullptr, nullptr, q16, n, d, d, 0);
    wmma_gemm_tiled<<<gdd, 256, 0, stream>>>(h1, WkT, bk, nullptr, nullptr, k16, n, d, d, 0);
    wmma_gemm_tiled<<<gdd, 256, 0, stream>>>(h1, WvT, bv, nullptr, nullptr, v16, n, d, d, 0);

    // V^T (d x n) for K-contiguous P@V operands
    transpose_f16<<<(unsigned)((wnd + T - 1) / T), T, 0, stream>>>(v16, vTT, n, d);

    // causal flash attention -> ao16
    flash_attn<<<dim3(n / 16, HEADS), 32, 0, stream>>>(q16, k16, vTT, ao16, n, d);

    // out-proj + residual -> x1 (f32)
    wmma_gemm_tiled<<<gdd, 256, 0, stream>>>(ao16, WoT, bo, x, x1, nullptr, n, d, d, 0);

    // LN2 -> h2 (f16)
    ln_f16_kernel<<<n, 256, 0, stream>>>(x1, ln2_w, ln2_b, h2, d);

    // MLP up + GELU -> m16 (f16)
    wmma_gemm_tiled<<<dim3(dff / 128, n / 128), 256, 0, stream>>>(h2, WiT, bi, nullptr, nullptr, m16, n, dff, d, 1);

    // MLP down + bias + residual -> d_out (f32)
    wmma_gemm_tiled<<<dim3(d / 128, n / 128), 256, 0, stream>>>(m16, WoutT, bout, x1, (float*)d_out, nullptr, n, d, dff, 0);
}
